// LorentzAttention_54434415509618
// MI455X (gfx1250) — matrix operand order repaired and checked by reading, not compiled, not run on hardware
//
#include <hip/hip_runtime.h>
#include <hip/hip_bf16.h>

typedef __bf16 bf16;
typedef __attribute__((ext_vector_type(16))) __bf16 v16bf;
typedef __attribute__((ext_vector_type(8)))  __bf16 v8bf;
typedef __attribute__((ext_vector_type(8)))  float  v8f;
typedef __attribute__((ext_vector_type(4)))  unsigned int u32x4;
typedef __attribute__((ext_vector_type(4)))  int i32x4;
typedef __attribute__((ext_vector_type(8)))  int i32x8;

#define B_      4
#define N_      2048
#define SP_     512
#define H_      8
#define DH_     64
#define M_      (B_ * N_)          /* 8192 rows */
#define INV_K   10.0f
#define SQRT_K  0.31622776601683794f
#define SCALE_  0.125f             /* 1/sqrt(64) */

// ---------- WMMA fragment helpers (16-bit operands, wave32) ----------
// A-operand (16x32): lane half ls picks contiguous K {ls*16 .. ls*16+15} of its row.
static __device__ __forceinline__ v16bf make_afrag(const bf16* p) {
  v8bf lo = *(const v8bf*)p;
  v8bf hi = *(const v8bf*)(p + 8);
  return __builtin_shufflevector(lo, hi, 0,1,2,3,4,5,6,7,8,9,10,11,12,13,14,15);
}
// B-operand (32x16): lane half ls picks contiguous K {ls*16 .. ls*16+15} of its column.
static __device__ __forceinline__ v16bf make_bfrag(const bf16* p) {
  return *(const v16bf*)p;   // 32B-aligned contiguous
}
// wave32 xor-butterfly via ds_bpermute (no LDS memory traffic)
static __device__ __forceinline__ float xor_lane_f(float v, int mask) {
  int lane = (int)(threadIdx.x & 31);
  return __int_as_float(
      __builtin_amdgcn_ds_bpermute(((lane ^ mask) << 2), __float_as_int(v)));
}

// ---------- Tensor Data Mover: 2D bf16 tile  global -> LDS ----------
// D# per CDNA5 ISA ch.8: group0 = {count|flags, lds_addr, global_addr, type=2},
// group1 = {data_size=2B, tensor dims == tile dims (always in-bounds),
//           tile_dim0/1, tensor_dim0_stride}.  Groups 2/3 zero (2-D tile).
static __device__ __forceinline__ void tdm_load_2d(unsigned int lds_off,
                                                   const void* gptr,
                                                   unsigned int tile_x,   // elems (bf16)
                                                   unsigned int tile_y,   // rows
                                                   unsigned int stride)   // elems
{
  unsigned long long ga = (unsigned long long)(size_t)gptr;
  u32x4 g0;
  g0[0] = 1u;                                  // count=1 valid user descriptor
  g0[1] = lds_off;                             // LDS byte offset (tile stored packed)
  g0[2] = (unsigned int)ga;                    // global_addr[31:0]
  g0[3] = (unsigned int)(ga >> 32) | 0x80000000u;  // global_addr[56:32] | type=2<<30
  i32x8 g1;
  g1[0] = (int)(1u << 16);                     // data_size = 1 -> 2 bytes/elem
  g1[1] = (int)((tile_x & 0xFFFFu) << 16);     // tensor_dim0[15:0]
  g1[2] = (int)((tile_x >> 16) | ((tile_y & 0xFFFFu) << 16));  // td0 hi | td1 lo
  g1[3] = (int)((tile_y >> 16) | (tile_x << 16));              // td1 hi | tile_dim0
  g1[4] = (int)(tile_y & 0xFFFFu);             // tile_dim1 (tile_dim2 = 0)
  g1[5] = (int)stride;                         // tensor_dim0_stride[31:0]
  g1[6] = 0;
  g1[7] = 0;
  i32x4 z4 = {};
#if defined(__clang_major__) && (__clang_major__ >= 23)
  i32x8 z8 = {};
  __builtin_amdgcn_tensor_load_to_lds(g0, g1, z4, z4, z8, 0);
#else
  __builtin_amdgcn_tensor_load_to_lds(g0, g1, z4, z4, 0);
#endif
}
static __device__ __forceinline__ unsigned int lds_offset_of(const void* p) {
  return (unsigned int)(size_t)p;   // generic LDS pointer: low 32 bits = LDS offset
}

// ---------- one-time converts ----------
__global__ __launch_bounds__(256) void prep_x(const float* __restrict__ x,
                                              bf16* __restrict__ xsp,
                                              float* __restrict__ xt) {
  int row = blockIdx.x;
  const float* xr = x + row * 513;
#pragma unroll
  for (int i = 0; i < 2; i++) {
    int j = threadIdx.x + i * 256;
    xsp[row * 512 + j] = (bf16)xr[1 + j];
  }
  if (threadIdx.x == 0) xt[row] = xr[0];
}

// Wt[n][k] = W[(k+1)*512 + n]  (drop time row 0, transpose, bf16)
__global__ __launch_bounds__(256) void prep_w(const float* __restrict__ W,
                                              bf16* __restrict__ Wt) {
  int n = blockIdx.x;
#pragma unroll
  for (int i = 0; i < 2; i++) {
    int k = threadIdx.x + i * 256;
    Wt[n * 512 + k] = (bf16)W[(k + 1) * 512 + n];
  }
}

// ---------- WMMA GEMM (TDM double-buffered): C = A*W + xt⊗w0 + bias ----------
__global__ __launch_bounds__(256) void gemm_bf16(const bf16* __restrict__ A,
                                                 const bf16* __restrict__ Bt,  // [n*512+k]
                                                 const float* __restrict__ xt,
                                                 const float* __restrict__ w0,
                                                 const float* __restrict__ bias,
                                                 float* __restrict__ C) {
  __shared__ __align__(32) bf16 As[2][128 * 32];
  __shared__ __align__(32) bf16 Bs[2][64 * 32];
  const int tid  = threadIdx.x;
  const int wave = tid >> 5;
  const int lane = tid & 31;
  const int lh = lane & 15, ls = lane >> 4;
  const int m0 = blockIdx.y * 128;
  const int n0 = blockIdx.x * 64;
  const int wm = wave >> 1, wn = wave & 1;   // 4x2 wave grid, 32x32 per wave
  v8f acc[2][2] = {};

  const bf16* Agbl = A + (size_t)m0 * 512;
  const bf16* Bgbl = Bt + (size_t)n0 * 512;
  unsigned int aoff[2] = { lds_offset_of(&As[0][0]), lds_offset_of(&As[1][0]) };
  unsigned int boff[2] = { lds_offset_of(&Bs[0][0]), lds_offset_of(&Bs[1][0]) };

  if (wave == 0) {   // prologue: DMA first K-slab into buffer 0
    tdm_load_2d(aoff[0], Agbl, 32u, 128u, 512u);
    tdm_load_2d(boff[0], Bgbl, 32u, 64u, 512u);
  }

  for (int it = 0; it < 16; it++) {
    __builtin_amdgcn_s_wait_tensorcnt(0);   // current buffer's DMA done (wave0)
    __syncthreads();                        // publish LDS + close previous compute
    if (wave == 0 && it + 1 < 16) {         // DMA next slab into other buffer
      int nb = (it + 1) & 1;
      tdm_load_2d(aoff[nb], Agbl + (it + 1) * 32, 32u, 128u, 512u);
      tdm_load_2d(boff[nb], Bgbl + (it + 1) * 32, 32u, 64u, 512u);
    }
    const bf16* Ab = &As[it & 1][0];
    const bf16* Bb = &Bs[it & 1][0];
    v16bf af[2], bfr[2];
#pragma unroll
    for (int mi = 0; mi < 2; mi++)
      af[mi] = make_afrag(Ab + (wm * 32 + mi * 16 + lh) * 32 + ls * 16);
#pragma unroll
    for (int ni = 0; ni < 2; ni++)
      bfr[ni] = make_bfrag(Bb + (wn * 32 + ni * 16 + lh) * 32 + ls * 16);
#pragma unroll
    for (int mi = 0; mi < 2; mi++)
#pragma unroll
      for (int ni = 0; ni < 2; ni++)
        acc[mi][ni] = __builtin_amdgcn_wmma_f32_16x16x32_bf16(
            false, af[mi], false, bfr[ni], (short)0, acc[mi][ni], false, false);
  }
#pragma unroll
  for (int mi = 0; mi < 2; mi++)
#pragma unroll
    for (int ni = 0; ni < 2; ni++)
#pragma unroll
      for (int v = 0; v < 8; v++) {
        int m = m0 + wm * 32 + mi * 16 + ls * 8 + v;
        int n = n0 + wn * 32 + ni * 16 + lh;
        C[(size_t)m * 512 + n] = acc[mi][ni][v] + xt[m] * w0[n] + bias[n];
      }
}

// ---------- q/k epilogue: bf16 copy + Lorentz time ----------
__global__ __launch_bounds__(256) void epi_qk(const float* __restrict__ C,
                                              bf16* __restrict__ sp_bf,
                                              float* __restrict__ t_out) {
  __shared__ float red[256];
  int row = blockIdx.x, tid = threadIdx.x;
  float s = 0.f;
#pragma unroll
  for (int i = 0; i < 2; i++) {
    float v = C[(size_t)row * 512 + tid + i * 256];
    s += v * v;
    sp_bf[(size_t)row * 512 + tid + i * 256] = (bf16)v;
  }
  red[tid] = s;
  __syncthreads();
  for (int o = 128; o > 0; o >>= 1) {
    if (tid < o) red[tid] += red[tid + o];
    __syncthreads();
  }
  if (tid == 0) t_out[row] = sqrtf(fmaxf(INV_K + red[0], 1e-9f));
}

// ---------- v epilogue: logmap0 per head -> bf16 ----------
__global__ __launch_bounds__(256) void epi_v(const float* __restrict__ C,
                                             bf16* __restrict__ vtan_bf) {
  __shared__ float hsum[8];
  int row = blockIdx.x, tid = threadIdx.x;
  int head = tid >> 5;  // one wave per head
  float2 u = *(const float2*)(C + (size_t)row * 512 + tid * 2);
  float hs = u.x * u.x + u.y * u.y;
#pragma unroll
  for (int m = 16; m > 0; m >>= 1) hs += xor_lane_f(hs, m);
  if ((tid & 31) == 0) hsum[head] = hs;
  __syncthreads();
  float tot = 0.f;
#pragma unroll
  for (int h = 0; h < 8; h++) tot += hsum[h];
  float t = sqrtf(INV_K + tot);
  float xx = fmaxf(SQRT_K * t, 1.0f + 1e-7f);
  float theta = __logf(xx + sqrtf(xx * xx - 1.f));   // arccosh
  float nrm = sqrtf(hs);
  float fac = theta / (SQRT_K * fmaxf(nrm, 1e-9f));
  bf16* o = vtan_bf + (size_t)row * 512 + tid * 2;
  o[0] = (bf16)(u.x * fac);
  o[1] = (bf16)(u.y * fac);
}

// ---------- flash attention per (b,h): TDM K tiles, WMMA S and P*V ----------
__global__ __launch_bounds__(256) void flash_attn(const bf16* __restrict__ qbf,
                                                  const bf16* __restrict__ kbf,
                                                  const bf16* __restrict__ vbf,
                                                  const float* __restrict__ qt,
                                                  const float* __restrict__ kt,
                                                  float* __restrict__ outp) {
  __shared__ __align__(32) bf16 Ks[64 * 64];       // [key][d]  (TDM-filled)
  __shared__ __align__(32) bf16 Vt[64 * 64];       // [d][key]
  __shared__ __align__(32) bf16 Ps[8 * 16 * 64];   // per-wave P stage
  const int tid  = threadIdx.x;
  const int wave = tid >> 5;
  const int lane = tid & 31;
  const int lh = lane & 15, ls = lane >> 4;
  const int bh = blockIdx.y;
  const int b = bh >> 3, h = bh & 7;
  const int hc = h * DH_;
  const int qrow = b * N_ + blockIdx.x * 128 + wave * 16;
  const unsigned int ksoff = lds_offset_of(&Ks[0]);

  v16bf qa[2];
#pragma unroll
  for (int kstep = 0; kstep < 2; kstep++)
    qa[kstep] = make_afrag(qbf + (size_t)(qrow + lh) * 512 + hc + kstep * 32 + ls * 16);
  float qtr[8];
#pragma unroll
  for (int v = 0; v < 8; v++) qtr[v] = qt[qrow + ls * 8 + v];

  float mst[8], lst[8];
  v8f oacc[4] = {};
#pragma unroll
  for (int v = 0; v < 8; v++) { mst[v] = -3.0e38f; lst[v] = 0.f; }
  bf16* Pw = Ps + wave * (16 * 64);

  for (int j = 0; j < 32; j++) {
    const int kr = b * N_ + j * 64;
    if (wave == 0)   // DMA K tile [key][d] (row-contiguous 64x64 bf16)
      tdm_load_2d(ksoff, kbf + (size_t)kr * 512 + hc, 64u, 64u, 512u);
    {  // stage V tile transposed [d][key] (overlaps K DMA)
      int r = tid >> 2, c = (tid & 3) * 16;
      v8bf a0 = *(const v8bf*)(vbf + (size_t)(kr + r) * 512 + hc + c);
      v8bf a1 = *(const v8bf*)(vbf + (size_t)(kr + r) * 512 + hc + c + 8);
#pragma unroll
      for (int i = 0; i < 8; i++) {
        Vt[(c + i) * 64 + r] = a0[i];
        Vt[(c + 8 + i) * 64 + r] = a1[i];
      }
    }
    __builtin_amdgcn_s_wait_tensorcnt(0);
    __syncthreads();

    // S = Q * K^T  (16x64 per wave)
    v8f sacc[4] = {};
#pragma unroll
    for (int kstep = 0; kstep < 2; kstep++)
#pragma unroll
      for (int nt = 0; nt < 4; nt++) {
        v16bf kfr = make_bfrag(Ks + (nt * 16 + lh) * 64 + kstep * 32 + ls * 16);
        sacc[nt] = __builtin_amdgcn_wmma_f32_16x16x32_bf16(
            false, qa[kstep], false, kfr, (short)0, sacc[nt], false, false);
      }
    // rank-1 time correction + scale
    float ktv[4];
#pragma unroll
    for (int nt = 0; nt < 4; nt++) ktv[nt] = kt[kr + nt * 16 + lh];
#pragma unroll
    for (int nt = 0; nt < 4; nt++)
#pragma unroll
      for (int v = 0; v < 8; v++)
        sacc[nt][v] = (sacc[nt][v] - qtr[v] * ktv[nt]) * SCALE_;

    // online softmax (row = 16 lanes within a half-wave)
#pragma unroll
    for (int v = 0; v < 8; v++) {
      float mx = fmaxf(fmaxf(sacc[0][v], sacc[1][v]), fmaxf(sacc[2][v], sacc[3][v]));
#pragma unroll
      for (int m = 8; m > 0; m >>= 1) mx = fmaxf(mx, xor_lane_f(mx, m));
      float mnew = fmaxf(mst[v], mx);
      float msc = __expf(mst[v] - mnew);
      float rs = 0.f;
#pragma unroll
      for (int nt = 0; nt < 4; nt++) {
        float p = __expf(sacc[nt][v] - mnew);
        sacc[nt][v] = p;
        rs += p;
      }
#pragma unroll
      for (int m = 8; m > 0; m >>= 1) rs += xor_lane_f(rs, m);
      lst[v] = lst[v] * msc + rs;
      mst[v] = mnew;
#pragma unroll
      for (int d = 0; d < 4; d++) oacc[d][v] *= msc;
    }

    // stage P (A-operand layout source) in this wave's LDS slice
#pragma unroll
    for (int nt = 0; nt < 4; nt++)
#pragma unroll
      for (int v = 0; v < 8; v++)
        Pw[(ls * 8 + v) * 64 + nt * 16 + lh] = (bf16)sacc[nt][v];

    // O += P * V
#pragma unroll
    for (int kstep = 0; kstep < 2; kstep++) {
      v16bf pa = make_afrag(Pw + lh * 64 + kstep * 32 + ls * 16);
#pragma unroll
      for (int d = 0; d < 4; d++) {
        v16bf vfr = make_bfrag(Vt + (d * 16 + lh) * 64 + kstep * 32 + ls * 16);
        oacc[d] = __builtin_amdgcn_wmma_f32_16x16x32_bf16(
            false, pa, false, vfr, (short)0, oacc[d], false, false);
      }
    }
    __syncthreads();
  }

#pragma unroll
  for (int d = 0; d < 4; d++)
#pragma unroll
    for (int v = 0; v < 8; v++) {
      float o = oacc[d][v] / lst[v];
      outp[(size_t)(qrow + ls * 8 + v) * 512 + hc + d * 16 + lh] = o;
    }
}

// ---------- expmap0 per head + concat time ----------
__global__ __launch_bounds__(256) void epi_expmap(const float* __restrict__ C,
                                                  bf16* __restrict__ cat_bf,
                                                  float* __restrict__ tp) {
  __shared__ float red[256];
  int row = blockIdx.x, tid = threadIdx.x;
  float2 u = *(const float2*)(C + (size_t)row * 512 + tid * 2);
  float hs = u.x * u.x + u.y * u.y;
#pragma unroll
  for (int m = 16; m > 0; m >>= 1) hs += xor_lane_f(hs, m);
  float nrm = sqrtf(hs);
  float a = SQRT_K * nrm;
  float ea = __expf(a);
  float sh = 0.5f * (ea - 1.f / ea);
  float fac = sh / (SQRT_K * fmaxf(nrm, 1e-9f));
  float s0 = u.x * fac, s1 = u.y * fac;
  bf16* o = cat_bf + (size_t)row * 512 + tid * 2;
  o[0] = (bf16)s0;
  o[1] = (bf16)s1;
  red[tid] = s0 * s0 + s1 * s1;
  __syncthreads();
  for (int o2 = 128; o2 > 0; o2 >>= 1) {
    if (tid < o2) red[tid] += red[tid + o2];
    __syncthreads();
  }
  if (tid == 0) tp[row] = sqrtf(fmaxf(INV_K + red[0], 1e-9f));
}

// ---------- final: time + pack [t, sp] (B,N,513) ----------
__global__ __launch_bounds__(256) void epi_final(const float* __restrict__ C,
                                                 float* __restrict__ out) {
  __shared__ float red[256];
  int row = blockIdx.x, tid = threadIdx.x;
  float v0 = C[(size_t)row * 512 + tid];
  float v1 = C[(size_t)row * 512 + tid + 256];
  red[tid] = v0 * v0 + v1 * v1;
  __syncthreads();
  for (int o = 128; o > 0; o >>= 1) {
    if (tid < o) red[tid] += red[tid + o];
    __syncthreads();
  }
  float* orow = out + (size_t)row * 513;
  if (tid == 0) orow[0] = sqrtf(fmaxf(INV_K + red[0], 1e-9f));
  orow[1 + tid] = v0;
  orow[1 + tid + 256] = v1;
}

extern "C" void kernel_launch(void* const* d_in, const int* in_sizes, int n_in,
                              void* d_out, int out_size, void* d_ws, size_t ws_size,
                              hipStream_t stream) {
  (void)in_sizes; (void)n_in; (void)out_size; (void)ws_size;
  const float* x  = (const float*)d_in[0];
  const float* Wq = (const float*)d_in[1];
  const float* bq = (const float*)d_in[2];
  const float* Wk = (const float*)d_in[3];
  const float* bk = (const float*)d_in[4];
  const float* Wv = (const float*)d_in[5];
  const float* bv = (const float*)d_in[6];
  const float* Wo = (const float*)d_in[7];
  const float* bo = (const float*)d_in[8];
  float* out = (float*)d_out;
  char* ws = (char*)d_ws;

  size_t off = 0;
  bf16*  xsp = (bf16*)(ws + off);  off += (size_t)M_ * 512 * 2;   // reused as cat_bf
  float* xt  = (float*)(ws + off); off += (size_t)M_ * 4;         // reused as t'
  bf16*  wqt = (bf16*)(ws + off);  off += 512 * 512 * 2;
  bf16*  wkt = (bf16*)(ws + off);  off += 512 * 512 * 2;
  bf16*  wvt = (bf16*)(ws + off);  off += 512 * 512 * 2;
  bf16*  wot = (bf16*)(ws + off);  off += 512 * 512 * 2;
  bf16*  qb  = (bf16*)(ws + off);  off += (size_t)M_ * 512 * 2;
  bf16*  kb  = (bf16*)(ws + off);  off += (size_t)M_ * 512 * 2;
  bf16*  vb  = (bf16*)(ws + off);  off += (size_t)M_ * 512 * 2;
  float* qt  = (float*)(ws + off); off += (size_t)M_ * 4;
  float* kt  = (float*)(ws + off); off += (size_t)M_ * 4;
  float* scr = (float*)(ws + off); off += (size_t)M_ * 512 * 4;   // f32 scratch (reused)

  prep_x<<<M_, 256, 0, stream>>>(x, xsp, xt);
  prep_w<<<512, 256, 0, stream>>>(Wq, wqt);
  prep_w<<<512, 256, 0, stream>>>(Wk, wkt);
  prep_w<<<512, 256, 0, stream>>>(Wv, wvt);
  prep_w<<<512, 256, 0, stream>>>(Wo, wot);

  dim3 gg(8, 64);  // 512/64 x 8192/128
  gemm_bf16<<<gg, 256, 0, stream>>>(xsp, wqt, xt, Wq, bq, scr);
  epi_qk<<<M_, 256, 0, stream>>>(scr, qb, qt);
  gemm_bf16<<<gg, 256, 0, stream>>>(xsp, wkt, xt, Wk, bk, scr);
  epi_qk<<<M_, 256, 0, stream>>>(scr, kb, kt);
  gemm_bf16<<<gg, 256, 0, stream>>>(xsp, wvt, xt, Wv, bv, scr);
  epi_v<<<M_, 256, 0, stream>>>(scr, vb);

  dim3 ga(16, 32);  // 2048/128 query blocks x (B*H) heads
  flash_attn<<<ga, 256, 0, stream>>>(qb, kb, vb, qt, kt, scr);

  epi_expmap<<<M_, 256, 0, stream>>>(scr, xsp, xt);
  gemm_bf16<<<gg, 256, 0, stream>>>(xsp, wot, xt, Wo, bo, scr);
  epi_final<<<M_, 256, 0, stream>>>(scr, out);
}
